// RecurrentNeuralBeamformerTeacherForcing_68135361183986
// MI455X (gfx1250) — matrix operand order repaired and checked by reading, not compiled, not run on hardware
//
#include <hip/hip_runtime.h>
#include <stdint.h>

// ---------------------------------------------------------------------------
// RecurrentNeuralBeamformerTeacherForcing for MI455X (gfx1250)
//
// Phase 1: per-(n,f) 8x8 Hermitian SCM + Gauss-Jordan inverse  (tiny)
// Phase 2: rank-1 MCWF:  Ybf_d = xprev_d * (y^T conj(invPhi) conj(xprev))
//          + build bf16 "frames" inputs for both TDU nets
// Phase 3: 40 dilated conv1d layers + 1x1 comb as bf16 WMMA GEMMs
//          (v_wmma_f32_16x16x32_bf16), M=out-ch tile 16, N=freq tile 16,
//          K=in-ch chunks of 32; taps = 3 shifted GEMM accumulations.
//          Layer shapes are template parameters so the K-loop fully unrolls.
//          LDS staging uses gfx1250 async global->LDS DMA when available.
// ---------------------------------------------------------------------------

#define FQ   257
#define TT   128
#define NN   2
#define CCH  8
#define KOUT 80
#define NFR  (NN * TT)          // 256 frames
#define FTILES ((FQ + 15) / 16) // 17

typedef __attribute__((ext_vector_type(16))) __bf16 v16bf;
typedef __attribute__((ext_vector_type(8)))  float  v8f;

#if defined(__has_builtin)
#if __has_builtin(__builtin_amdgcn_global_load_async_to_lds_b128)
#define USE_ASYNC_LDS 1
#endif
#endif

#ifdef USE_ASYNC_LDS
typedef int v4i_ __attribute__((ext_vector_type(4)));
typedef __attribute__((address_space(1))) void as1_void;
typedef __attribute__((address_space(3))) void as3_void;
typedef __attribute__((address_space(1))) v4i_ as1_v4i;
typedef __attribute__((address_space(3))) v4i_ as3_v4i;
static __device__ __forceinline__ void wait_asynccnt0() {
#if __has_builtin(__builtin_amdgcn_s_wait_asynccnt)
  __builtin_amdgcn_s_wait_asynccnt(0);
#else
  asm volatile("s_wait_asynccnt 0" ::: "memory");
#endif
}
#endif

static __device__ __forceinline__ unsigned short f2bf(float x) {
  union { float f; unsigned u; } v; v.f = x;
  unsigned r = (v.u + 0x7FFFu + ((v.u >> 16) & 1u)) >> 16;
  return (unsigned short)r;
}
static __device__ __forceinline__ float bf2f(unsigned short b) {
  union { unsigned u; float f; } v; v.u = ((unsigned)b) << 16; return v.f;
}

// ---------------------------------------------------------------------------
// Phase 1: inv(phi_Y + DIAG*I) per (n,f).  514 threads, one 8x8 complex GJ each.
// invPhi layout: [n][f][c][d][2] float
// ---------------------------------------------------------------------------
__global__ void invphi_kernel(const float* __restrict__ Y, float* __restrict__ invPhi) {
  int t = blockIdx.x * blockDim.x + threadIdx.x;
  if (t >= NN * FQ) return;
  int n = t / FQ, f = t % FQ;

  const float* Yr = Y + (size_t)(n * 2 + 0) * CCH * TT * FQ;
  const float* Yi = Y + (size_t)(n * 2 + 1) * CCH * TT * FQ;

  float ar[8][8], ai[8][8], br[8][8], bi[8][8];
  for (int c = 0; c < 8; ++c)
    for (int d = 0; d < 8; ++d) { ar[c][d] = 0.f; ai[c][d] = 0.f; br[c][d] = 0.f; bi[c][d] = 0.f; }

  for (int tt = 0; tt < TT; ++tt) {
    float yr[8], yi[8];
    for (int c = 0; c < 8; ++c) {
      yr[c] = Yr[((size_t)c * TT + tt) * FQ + f];
      yi[c] = Yi[((size_t)c * TT + tt) * FQ + f];
    }
    for (int c = 0; c < 8; ++c)
      for (int d = 0; d < 8; ++d) {
        ar[c][d] += yr[c] * yr[d] + yi[c] * yi[d];   // y_c * conj(y_d)
        ai[c][d] += yi[c] * yr[d] - yr[c] * yi[d];
      }
  }
  const float sc = 1.0f / (float)TT;
  for (int c = 0; c < 8; ++c)
    for (int d = 0; d < 8; ++d) { ar[c][d] *= sc; ai[c][d] *= sc; }
  for (int c = 0; c < 8; ++c) { ar[c][c] += 1e-4f; br[c][c] = 1.f; }

  // complex Gauss-Jordan (Hermitian PD -> no pivoting)
  for (int k = 0; k < 8; ++k) {
    float pr = ar[k][k], pi = ai[k][k];
    float den = 1.f / (pr * pr + pi * pi);
    float qr = pr * den, qi = -pi * den;
    for (int j = 0; j < 8; ++j) {
      float xr = ar[k][j], xi = ai[k][j];
      ar[k][j] = xr * qr - xi * qi; ai[k][j] = xr * qi + xi * qr;
      xr = br[k][j]; xi = bi[k][j];
      br[k][j] = xr * qr - xi * qi; bi[k][j] = xr * qi + xi * qr;
    }
    for (int i = 0; i < 8; ++i) {
      if (i == k) continue;
      float fr = ar[i][k], fi = ai[i][k];
      for (int j = 0; j < 8; ++j) {
        ar[i][j] -= fr * ar[k][j] - fi * ai[k][j];
        ai[i][j] -= fr * ai[k][j] + fi * ar[k][j];
        br[i][j] -= fr * br[k][j] - fi * bi[k][j];
        bi[i][j] -= fr * bi[k][j] + fi * br[k][j];
      }
    }
  }
  float* dst = invPhi + (size_t)t * 128;
  for (int c = 0; c < 8; ++c)
    for (int d = 0; d < 8; ++d) {
      dst[(c * 8 + d) * 2 + 0] = br[c][d];
      dst[(c * 8 + d) * 2 + 1] = bi[c][d];
    }
}

// ---------------------------------------------------------------------------
// Phase 2: rank-1 beamformer + frames (bf16), one thread per (n,t,f)
// framesY/framesB: [frame][f][16] bf16 (real ch 0..7, imag ch 8..15)
// ---------------------------------------------------------------------------
__global__ void beamform_frames_kernel(const float* __restrict__ Y,
                                       const float* __restrict__ X,
                                       const float* __restrict__ invPhi,
                                       unsigned short* __restrict__ framesY,
                                       unsigned short* __restrict__ framesB) {
  int t = blockIdx.x * blockDim.x + threadIdx.x;
  if (t >= NN * TT * FQ) return;
  int f = t % FQ, r = t / FQ;
  int tt = r % TT, n = r / TT;
  int frame = n * TT + tt;

  float yr[8], yi[8], xr[8], xi[8];
  const float* Yrp = Y + (size_t)(n * 2 + 0) * CCH * TT * FQ;
  const float* Yip = Y + (size_t)(n * 2 + 1) * CCH * TT * FQ;
  const float* Xrp = X + (size_t)(n * 2 + 0) * CCH * TT * FQ;
  const float* Xip = X + (size_t)(n * 2 + 1) * CCH * TT * FQ;
  for (int c = 0; c < 8; ++c) {
    yr[c] = Yrp[((size_t)c * TT + tt) * FQ + f];
    yi[c] = Yip[((size_t)c * TT + tt) * FQ + f];
    if (tt == 0) { xr[c] = 0.f; xi[c] = 0.f; }
    else {
      xr[c] = Xrp[((size_t)c * TT + (tt - 1)) * FQ + f];
      xi[c] = Xip[((size_t)c * TT + (tt - 1)) * FQ + f];
    }
  }
  unsigned short* fy = framesY + ((size_t)frame * FQ + f) * 16;
  for (int c = 0; c < 8; ++c) { fy[c] = f2bf(yr[c]); fy[8 + c] = f2bf(yi[c]); }

  // s = sum_c y_c * sum_e conj(P[c][e]) * conj(xp_e)
  const float* P = invPhi + ((size_t)n * FQ + f) * 128;
  float sr = 0.f, si = 0.f;
  for (int c = 0; c < 8; ++c) {
    float ur = 0.f, ui = 0.f;
    for (int e = 0; e < 8; ++e) {
      float pr = P[(c * 8 + e) * 2 + 0], pi = P[(c * 8 + e) * 2 + 1];
      ur += pr * xr[e] - pi * xi[e];
      ui += -(pr * xi[e] + pi * xr[e]);
    }
    sr += yr[c] * ur - yi[c] * ui;
    si += yr[c] * ui + yi[c] * ur;
  }
  unsigned short* fb = framesB + ((size_t)frame * FQ + f) * 16;
  for (int d = 0; d < 8; ++d) {
    float brv = xr[d] * sr - xi[d] * si;
    float biv = xr[d] * si + xi[d] * sr;
    fb[d] = f2bf(brv); fb[8 + d] = f2bf(biv);
  }
}

// ---------------------------------------------------------------------------
// Pack fp32 conv weights (Kout, CinTotal, ntaps) -> WMMA A-operand lane layout.
// Block = (mtile, src, tap, kchunk); 32 lanes x 16 bf16 each = 1 KB.
// Lane l: M = l&15; half h -> K = h + 16*(l>>4)
// ---------------------------------------------------------------------------
__global__ void pack_weights_kernel(const float* __restrict__ W,
                                    unsigned short* __restrict__ dst,
                                    int mtiles, int nsrc, int ntaps, int kch,
                                    int cinTotal, int cinSrc, int kout) {
  int nb = mtiles * nsrc * ntaps * kch;
  int t = blockIdx.x * blockDim.x + threadIdx.x;
  if (t >= nb * 32) return;
  int bid = t >> 5, lane = t & 31;
  int kc = bid % kch; int r = bid / kch;
  int tap = r % ntaps; r /= ntaps;
  int src = r % nsrc;  int mt = r / nsrc;
  int M = lane & 15, grp = lane >> 4;
  int oc = mt * 16 + M;
  unsigned short* o = dst + (size_t)bid * 512 + lane * 16;
  for (int h = 0; h < 16; ++h) {
    int K = h + 16 * grp;
    int ic = kc * 32 + K;
    float v = 0.f;
    if (oc < kout && ic < cinSrc)
      v = W[((size_t)oc * cinTotal + src * cinSrc + ic) * ntaps + tap];
    o[h] = f2bf(v);
  }
}

// ---------------------------------------------------------------------------
// Dilated conv1d layer as WMMA GEMM.  160 threads = 5 waves; wave = M-tile.
// Grid: (frames=256, ftiles=17).  LDS: [src][tap][16 f][96 ch] bf16.
// Template params make all GEMM loops compile-time so K-loop fully unrolls.
// Staging uses async global->LDS DMA (ASYNCcnt) when the builtin exists.
// ---------------------------------------------------------------------------
template <int CIN0, int CIN1, int NTAPS, int MTILES>
__global__ __launch_bounds__(160) void conv_wmma_kernel(
    const unsigned short* __restrict__ in0, const unsigned short* __restrict__ in1,
    const unsigned short* __restrict__ packA, const float* __restrict__ bias,
    unsigned short* __restrict__ out, int dil, int relu) {
  constexpr int NSRC = (CIN1 > 0) ? 2 : 1;
  constexpr int KCH  = (CIN0 <= 32) ? 1 : 3;
  constexpr int COUT = MTILES * 16;

  const int frame = blockIdx.x;
  const int f0 = blockIdx.y * 16;
  __shared__ unsigned short lds[2 * 3 * 16 * 96];

  // Vectorized staging: 8 bf16 (16 B) per iteration; all CINs are multiples of 8.
  constexpr int NCHUNK = 96 / 8; // 12
  constexpr int TOTAL = NSRC * NTAPS * 16 * NCHUNK;
  for (int e = threadIdx.x; e < TOTAL; e += 160) {
    int chk = e % NCHUNK; int r = e / NCHUNK;
    int frow = r % 16; r /= 16;
    int tap = r % NTAPS; int src = r / NTAPS;
    int gf = f0 + frow + (tap - NTAPS / 2) * dil;
    const unsigned short* sp = src ? in1 : in0;
    const int cinS = src ? CIN1 : CIN0;
    unsigned short* lp = &lds[((src * 3 + tap) * 16 + frow) * 96 + chk * 8];
    if (chk * 8 < cinS && gf >= 0 && gf < FQ) {
      const unsigned short* gp = &sp[((size_t)frame * FQ + gf) * cinS + chk * 8];
#ifdef USE_ASYNC_LDS
      __builtin_amdgcn_global_load_async_to_lds_b128(
          (as1_v4i*)(as1_void*)(void*)gp, (as3_v4i*)(as3_void*)(void*)lp, 0, 0);
#else
      *reinterpret_cast<uint4*>(lp) = *reinterpret_cast<const uint4*>(gp);
#endif
    } else {
      *reinterpret_cast<uint4*>(lp) = make_uint4(0u, 0u, 0u, 0u);
    }
  }
#ifdef USE_ASYNC_LDS
  wait_asynccnt0();
#endif
  __syncthreads();

  const int wave = threadIdx.x >> 5;
  const int lane = threadIdx.x & 31;
  if (wave >= MTILES) return;

  v8f acc = {0.f, 0.f, 0.f, 0.f, 0.f, 0.f, 0.f, 0.f};
  const unsigned short* aw = packA + (size_t)(wave * NSRC * NTAPS * KCH) * 512 + lane * 16;
  const unsigned short* lb = &lds[(lane & 15) * 96 + ((lane >> 4) * 16)];
#pragma unroll
  for (int src = 0; src < NSRC; ++src)
#pragma unroll
    for (int tap = 0; tap < NTAPS; ++tap)
#pragma unroll
      for (int kc = 0; kc < KCH; ++kc) {
        const int bid = (src * NTAPS + tap) * KCH + kc;
        const v16bf a = *reinterpret_cast<const v16bf*>(aw + (size_t)bid * 512);
        const v16bf b = *reinterpret_cast<const v16bf*>(
            lb + ((src * 3 + tap) * 16) * 96 + kc * 32);
        acc = __builtin_amdgcn_wmma_f32_16x16x32_bf16(
            /*neg_a=*/false, a, /*neg_b=*/false, b,
            /*c_mod=*/(short)0, acc, /*reuse_a=*/false, /*reuse_b=*/false);
      }

  const int f = f0 + (lane & 15);
  if (f < FQ) {
    const int ocb = wave * 16 + (lane >> 4) * 8;
    alignas(16) unsigned short o[8];
#pragma unroll
    for (int r = 0; r < 8; ++r) {
      float v = acc[r] + bias[ocb + r];
      if (relu) v = v > 0.f ? v : 0.f;
      o[r] = f2bf(v);
    }
    *reinterpret_cast<uint4*>(&out[((size_t)frame * FQ + f) * COUT + ocb]) =
        *reinterpret_cast<uint4*>(o);
  }
}

// ---------------------------------------------------------------------------
// Final reshape: comb [frame][f][16] bf16 -> d_out [N,2,C,T,FQ] fp32
// ---------------------------------------------------------------------------
__global__ void write_out_kernel(const unsigned short* __restrict__ comb,
                                 float* __restrict__ out) {
  size_t i = (size_t)blockIdx.x * blockDim.x + threadIdx.x;
  const size_t total = (size_t)NN * 2 * CCH * TT * FQ;
  if (i >= total) return;
  int f = (int)(i % FQ); size_t r = i / FQ;
  int tt = (int)(r % TT); r /= TT;
  int c = (int)(r % CCH); r /= CCH;
  int ri = (int)(r % 2);  int n = (int)(r / 2);
  out[i] = bf2f(comb[(((size_t)(n * TT + tt)) * FQ + f) * 16 + ri * CCH + c]);
}

// ---------------------------------------------------------------------------
extern "C" void kernel_launch(void* const* d_in, const int* in_sizes, int n_in,
                              void* d_out, int out_size, void* d_ws, size_t ws_size,
                              hipStream_t stream) {
  (void)in_sizes; (void)n_in; (void)out_size; (void)ws_size;
  const float* Y = (const float*)d_in[0];
  const float* X = (const float*)d_in[1];
  // params flattened in insertion order: enc_r(10x(w,b)), dec_r, enc_b, dec_b, comb_w, comb_b
  auto encW = [&](int net, int i) { return (const float*)d_in[2 + net * 40 + 2 * i]; };
  auto encB = [&](int net, int i) { return (const float*)d_in[3 + net * 40 + 2 * i]; };
  auto decW = [&](int net, int l) { return (const float*)d_in[22 + net * 40 + 2 * l]; };
  auto decB = [&](int net, int l) { return (const float*)d_in[23 + net * 40 + 2 * l]; };
  const float* combW = (const float*)d_in[82];
  const float* combB = (const float*)d_in[83];

  char* wsb = (char*)d_ws;
  size_t off = 0;
  auto alloc = [&](size_t bytes) -> char* {
    char* p = wsb + off; off = (off + bytes + 255) & ~(size_t)255; return p;
  };

  float* invPhi = (float*)alloc((size_t)NN * FQ * 64 * 2 * sizeof(float));
  unsigned short* framesY = (unsigned short*)alloc((size_t)NFR * FQ * 16 * 2);
  unsigned short* framesB = (unsigned short*)alloc((size_t)NFR * FQ * 16 * 2);
  const size_t actBytes = (size_t)NFR * FQ * KOUT * 2;
  unsigned short* encOut[2][10];
  for (int net = 0; net < 2; ++net)
    for (int i = 0; i < 10; ++i) encOut[net][i] = (unsigned short*)alloc(actBytes);
  unsigned short* decA[2], * decBuf[2];
  for (int net = 0; net < 2; ++net) {
    decA[net] = (unsigned short*)alloc(actBytes);
    decBuf[net] = (unsigned short*)alloc(actBytes);
  }
  unsigned short* combOut = (unsigned short*)alloc((size_t)NFR * FQ * 16 * 2);

  auto pack = [&](const float* W, int mtiles, int nsrc, int ntaps, int kch,
                  int cinTotal, int cinSrc, int kout) -> unsigned short* {
    int nb = mtiles * nsrc * ntaps * kch;
    unsigned short* dst = (unsigned short*)alloc((size_t)nb * 1024);
    int threads = nb * 32;
    pack_weights_kernel<<<(threads + 255) / 256, 256, 0, stream>>>(
        W, dst, mtiles, nsrc, ntaps, kch, cinTotal, cinSrc, kout);
    return dst;
  };

  unsigned short* packEnc[2][10];
  unsigned short* packDec[2][10];
  for (int net = 0; net < 2; ++net) {
    packEnc[net][0] = pack(encW(net, 0), 5, 1, 3, 1, 16, 16, 80);
    for (int i = 1; i < 10; ++i)
      packEnc[net][i] = pack(encW(net, i), 5, 1, 3, 3, 80, 80, 80);
    packDec[net][0] = pack(decW(net, 0), 5, 1, 3, 3, 80, 80, 80);
    for (int l = 1; l < 10; ++l)
      packDec[net][l] = pack(decW(net, l), 5, 2, 3, 3, 160, 80, 80);
  }
  unsigned short* packComb = pack(combW, 1, 2, 1, 3, 160, 80, 16);

  invphi_kernel<<<(NN * FQ + 127) / 128, 128, 0, stream>>>(Y, invPhi);
  beamform_frames_kernel<<<(NN * TT * FQ + 255) / 256, 256, 0, stream>>>(
      Y, X, invPhi, framesY, framesB);

  const dim3 grid(NFR, FTILES);
  unsigned short* netFinal[2];
  for (int net = 0; net < 2; ++net) {
    const unsigned short* frames = (net == 0) ? framesY : framesB;
    // encoder
    conv_wmma_kernel<16, 0, 3, 5><<<grid, 160, 0, stream>>>(
        frames, nullptr, packEnc[net][0], encB(net, 0), encOut[net][0], 1, 1);
    for (int i = 1; i < 10; ++i)
      conv_wmma_kernel<80, 0, 3, 5><<<grid, 160, 0, stream>>>(
          encOut[net][i - 1], nullptr, packEnc[net][i], encB(net, i),
          encOut[net][i], 1 << i, 1);
    // decoder (skip = encOut[9-l] for l>0)
    conv_wmma_kernel<80, 0, 3, 5><<<grid, 160, 0, stream>>>(
        encOut[net][9], nullptr, packDec[net][0], decB(net, 0), decA[net], 1 << 9, 1);
    const unsigned short* cur = decA[net];
    for (int l = 1; l < 10; ++l) {
      unsigned short* outb = (l & 1) ? decBuf[net] : decA[net];
      conv_wmma_kernel<80, 80, 3, 5><<<grid, 160, 0, stream>>>(
          cur, encOut[net][9 - l], packDec[net][l], decB(net, l),
          outb, 1 << (9 - l), 1);
      cur = outb;
    }
    netFinal[net] = (unsigned short*)cur;  // l=9 odd -> decBuf
  }

  // 1x1 combinator: cols 0..79 -> net r (from Yc frames), 80..159 -> net b
  conv_wmma_kernel<80, 80, 1, 1><<<grid, 160, 0, stream>>>(
      netFinal[0], netFinal[1], packComb, combB, combOut, 1, 0);

  const size_t total = (size_t)NN * 2 * CCH * TT * FQ;
  write_out_kernel<<<(unsigned)((total + 255) / 256), 256, 0, stream>>>(
      combOut, (float*)d_out);
}
